// The_fine_tune_network_20555713479066
// MI455X (gfx1250) — matrix operand order repaired and checked by reading, not compiled, not run on hardware
//
#include <hip/hip_runtime.h>

typedef _Float16 h16;
typedef __attribute__((ext_vector_type(16))) _Float16 v16h;
typedef __attribute__((ext_vector_type(8)))  _Float16 v8h;
typedef __attribute__((ext_vector_type(8)))  float    v8f;
typedef __attribute__((ext_vector_type(4)))  float    f4;

#define B_ROWS 8192
#define DIM    512
#define FEAT   1024
#define HID    256
#define TOPK   11
#define EPS_C  1e-6f

#define MROWS  64   // rows per mining workgroup (4 waves x 16)
#define MCOLS  64   // column tile
#define KT     64   // K-step (2 WMMA depths per barrier)
#define CROWS  128  // rows per classifier workgroup (8 waves x 16)

__device__ __forceinline__ unsigned hashu(unsigned x){
  x ^= x >> 16; x *= 0x7feb352dU; x ^= x >> 15; x *= 0x846ca68bU; x ^= x >> 16; return x;
}

// A fragment 16x32 f16 (ISA 7.12.2): lane m=l%16, kbase = l<16?0:8,
// VGPR0-3 = K[kb..kb+7], VGPR4-7 = K[kb+16..kb+23]
__device__ __forceinline__ v16h frag_a16(const h16* base, int stride){
  const int lane = threadIdx.x & 31;
  const h16* p = base + (lane & 15) * stride + ((lane < 16) ? 0 : 8);
  v16h a;
  #pragma unroll
  for (int i = 0; i < 8; ++i){ a[i] = p[i]; a[i + 8] = p[i + 16]; }
  return a;
}
// B fragment 32x16 f16: lane n=l%16, kbase = l<16?0:8, elems 0-7 = K[kb..kb+7], 8-15 = K[kb+16..kb+23]
__device__ __forceinline__ v16h frag_b16(const h16* base, int stride, int n0){
  const int lane = threadIdx.x & 31;
  const h16* p = base + ((lane < 16) ? 0 : 8) * stride + n0 + (lane & 15);
  v16h b;
  #pragma unroll
  for (int i = 0; i < 8; ++i){ b[i] = p[i * stride]; b[i + 8] = p[(i + 16) * stride]; }
  return b;
}

__global__ void k_zero(float* out){ if (threadIdx.x == 0) out[0] = 0.f; }

// pack-convert x (8192x1024 f32) -> f16 once; one block per row
__global__ void __launch_bounds__(256) k_cvt_x(const float* __restrict__ x,
                                               h16* __restrict__ xh){
  const size_t base = (size_t)blockIdx.x * FEAT + threadIdx.x * 4;
  f4 v = *(const f4*)(x + base);
  auto p0 = __builtin_amdgcn_cvt_pkrtz(v[0], v[1]);   // v_cvt_pk_f16_f32
  auto p1 = __builtin_amdgcn_cvt_pkrtz(v[2], v[3]);
  uint2 o; o.x = __builtin_bit_cast(unsigned, p0); o.y = __builtin_bit_cast(unsigned, p1);
  *(uint2*)(xh + base) = o;
}

// pack-convert W1 (1024x256 f32) -> f16 once
__global__ void __launch_bounds__(256) k_cvt_w1(const float* __restrict__ W1,
                                                h16* __restrict__ W1h){
  const size_t base = ((size_t)blockIdx.x * 256 + threadIdx.x) * 4;
  f4 v = *(const f4*)(W1 + base);
  auto p0 = __builtin_amdgcn_cvt_pkrtz(v[0], v[1]);
  auto p1 = __builtin_amdgcn_cvt_pkrtz(v[2], v[3]);
  uint2 o; o.x = __builtin_bit_cast(unsigned, p0); o.y = __builtin_bit_cast(unsigned, p1);
  *(uint2*)(W1h + base) = o;
}

// cconst[j] = sum(p_j^2) - 2*eps*sum(p_j)   (column part of the ranking key; f32 source)
__global__ void __launch_bounds__(256) k_colconst(const float* __restrict__ x,
                                                  float* __restrict__ cconst){
  const int wave = threadIdx.x >> 5, lane = threadIdx.x & 31;
  const int j = blockIdx.x * 8 + wave;
  const float* p = x + (size_t)j * FEAT + DIM;
  float s2 = 0.f, s1 = 0.f;
  for (int k = lane; k < DIM; k += 32){ float v = p[k]; s2 += v * v; s1 += v; }
  #pragma unroll
  for (int off = 16; off; off >>= 1){
    s2 += __shfl_down(s2, off, 32);
    s1 += __shfl_down(s1, off, 32);
  }
  if (lane == 0) cconst[j] = s2 - 2.f * EPS_C * s1;
}

// interleave halves of two row-dwords into two [k][n]-pair dwords
#define PK(a, b, e0, e1)                                                     \
  do {                                                                       \
    Bld32[kg + (e0)][np] = ((a) & 0xffffu) | ((b) << 16);                    \
    Bld32[kg + (e1)][np] = ((a) >> 16) | ((b) & 0xffff0000u);                \
  } while (0)

// Fused similarity-GEMM + per-row top-11 mining + negative selection.
__global__ void __launch_bounds__(128)
k_mine(const h16* __restrict__ xh, const float* __restrict__ cconst,
       int* __restrict__ negidx){
  __shared__ h16      Alds[MROWS][72];   // 64 rows x 64 k (rows 144B, 16B aligned)
  __shared__ unsigned Bld32[KT][36];     // 64 k x 64 n, f16 pairs packed in dwords
  __shared__ float    Slds[4][16][68];   // per-wave 16x64 score scratch
  __shared__ float    Clds[MCOLS];
  __shared__ float    Tv[4][16][2][TOPK];
  __shared__ int      Ti[4][16][2][TOPK];

  const int tid  = threadIdx.x;
  const int wave = tid >> 5, lane = tid & 31;
  const int rowbase = blockIdx.x * MROWS;

  float tv[TOPK]; int ti[TOPK];
  #pragma unroll
  for (int t = 0; t < TOPK; ++t){ tv[t] = -3.4e38f; ti[t] = 0; }

  const int srow  = lane >> 1;       // scan: 2 lanes per row
  const int shalf = lane & 1;
  const int grow  = rowbase + wave * 16 + srow;
  const h16* bb0 = (const h16*)&Bld32[0][0];
  const h16* bb1 = (const h16*)&Bld32[32][0];

  for (int ct = 0; ct < B_ROWS / MCOLS; ++ct){
    const int colbase = ct * MCOLS;
    if (tid < MCOLS){
      Clds[tid] = cconst[colbase + tid];
      if (ct + 1 < B_ROWS / MCOLS)   // global_prefetch_b8 of next P tile
        __builtin_prefetch(xh + (size_t)(colbase + MCOLS + tid) * FEAT + DIM, 0, 0);
    }

    v8f acc0 = {}, acc1 = {}, acc2 = {}, acc3 = {};
    for (int kb = 0; kb < DIM; kb += KT){
      __syncthreads();
      { // A tile: 64x64 halves, raw async copy global->LDS (512 x 16B chunks)
        #pragma unroll
        for (int q = 0; q < 4; ++q){
          const int c  = tid + q * 128;
          const int m  = c >> 3;
          const int ko = (c & 7) * 8;
          const unsigned lds = (unsigned)(size_t)&Alds[m][ko];
          const unsigned long long g =
            (unsigned long long)(size_t)(xh + (size_t)(rowbase + m) * FEAT + kb + ko);
          asm volatile("global_load_async_to_lds_b128 %0, %1, off"
                       :: "v"(lds), "v"(g) : "memory");
        }
      }
      { // B tile: transpose P rows -> [k][n]; two n per thread, explicit dword interleave
        const int np = tid & 31;           // n-pair id
        const int n0 = np * 2;
        const int kg = (tid >> 5) * 16;    // 0,16,32,48
        const h16* r0 = xh + (size_t)(colbase + n0) * FEAT + DIM + kb + kg;
        const uint4 x0 = *(const uint4*)r0;
        const uint4 x1 = *(const uint4*)(r0 + 8);
        const uint4 y0 = *(const uint4*)(r0 + FEAT);
        const uint4 y1 = *(const uint4*)(r0 + FEAT + 8);
        PK(x0.x, y0.x,  0,  1); PK(x0.y, y0.y,  2,  3);
        PK(x0.z, y0.z,  4,  5); PK(x0.w, y0.w,  6,  7);
        PK(x1.x, y1.x,  8,  9); PK(x1.y, y1.y, 10, 11);
        PK(x1.z, y1.z, 12, 13); PK(x1.w, y1.w, 14, 15);
      }
      asm volatile("s_wait_asynccnt 0" ::: "memory");
      __syncthreads();
      v16h af0 = frag_a16(&Alds[wave * 16][0], 72);
      v16h af1 = frag_a16(&Alds[wave * 16][32], 72);
      acc0 = __builtin_amdgcn_wmma_f32_16x16x32_f16(false, af0, false, frag_b16(bb0, 72,  0), (short)0, acc0, false, false);
      acc0 = __builtin_amdgcn_wmma_f32_16x16x32_f16(false, af1, false, frag_b16(bb1, 72,  0), (short)0, acc0, false, false);
      acc1 = __builtin_amdgcn_wmma_f32_16x16x32_f16(false, af0, false, frag_b16(bb0, 72, 16), (short)0, acc1, false, false);
      acc1 = __builtin_amdgcn_wmma_f32_16x16x32_f16(false, af1, false, frag_b16(bb1, 72, 16), (short)0, acc1, false, false);
      acc2 = __builtin_amdgcn_wmma_f32_16x16x32_f16(false, af0, false, frag_b16(bb0, 72, 32), (short)0, acc2, false, false);
      acc2 = __builtin_amdgcn_wmma_f32_16x16x32_f16(false, af1, false, frag_b16(bb1, 72, 32), (short)0, acc2, false, false);
      acc3 = __builtin_amdgcn_wmma_f32_16x16x32_f16(false, af0, false, frag_b16(bb0, 72, 48), (short)0, acc3, false, false);
      acc3 = __builtin_amdgcn_wmma_f32_16x16x32_f16(false, af1, false, frag_b16(bb1, 72, 48), (short)0, acc3, false, false);
    }
    { // scatter C fragments (row = r + 8*(lane>=16), col = lane%16)
      const int n = lane & 15, mo = (lane < 16) ? 0 : 8;
      #pragma unroll
      for (int r = 0; r < 8; ++r){
        Slds[wave][r + mo][ 0 + n] = acc0[r];
        Slds[wave][r + mo][16 + n] = acc1[r];
        Slds[wave][r + mo][32 + n] = acc2[r];
        Slds[wave][r + mo][48 + n] = acc3[r];
      }
    }
    __syncthreads();
    { // per-lane running top-11 over this column tile, vectorized loads
      const f4* sp4 = (const f4*)&Slds[wave][srow][shalf * 32];
      const f4* cp4 = (const f4*)&Clds[shalf * 32];
      #pragma unroll 2
      for (int i = 0; i < 8; ++i){
        f4 s = sp4[i], cc = cp4[i];
        #pragma unroll
        for (int e = 0; e < 4; ++e){
          const int gj = colbase + shalf * 32 + i * 4 + e;
          float key = 2.f * s[e] - cc[e];
          if (gj != grow && key > tv[TOPK - 1]){
            float cv = key; int ci = gj;
            #pragma unroll
            for (int t = 0; t < TOPK; ++t){
              const bool gt = cv > tv[t];
              const float ov = tv[t]; const int oi = ti[t];
              tv[t] = gt ? cv : ov;  ti[t] = gt ? ci : oi;
              cv    = gt ? ov : cv;  ci    = gt ? oi : ci;
            }
          }
        }
      }
    }
    __syncthreads();
  }

  #pragma unroll
  for (int t = 0; t < TOPK; ++t){
    Tv[wave][srow][shalf][t] = tv[t];
    Ti[wave][srow][shalf][t] = ti[t];
  }
  __syncthreads();

  if (tid < MROWS){
    const int w = tid >> 4, r = tid & 15;
    const int gi = rowbase + tid;
    const unsigned h1 = hashu((unsigned)gi * 2654435761u + 42u);
    const unsigned h2 = hashu(h1 + 0x9e3779b9u);
    const unsigned h3 = hashu(h2 + 0x85ebca6bu);
    const int hard_r = (int)(h1 % (unsigned)TOPK);
    const int rr = (int)(h2 % (unsigned)(B_ROWS - 1));
    const int rj = rr + (rr >= gi);           // uniform over non-self columns
    const bool use_hard = (h3 & 1u) != 0u;    // prob 0.5
    int ia = 0, ib = 0, chosen = 0;
    for (int s = 0; s < TOPK; ++s){           // merge two sorted 11-lists
      const float va = Tv[w][r][0][ia], vb = Tv[w][r][1][ib];
      const bool ta = va >= vb;
      const int id = ta ? Ti[w][r][0][ia] : Ti[w][r][1][ib];
      ia += ta; ib += !ta;
      if (s == hard_r) chosen = id;
    }
    negidx[gi] = use_hard ? chosen : rj;
  }
}

// Fused classifier: [feat(1024)]@W1 -> SiLU -> @W2 -> sigmoid -> BCE partial sum
__global__ void __launch_bounds__(256)
k_classify(const h16* __restrict__ xh, const h16* __restrict__ W1h,
           const float* __restrict__ b1, const float* __restrict__ W2,
           const float* __restrict__ b2, const int* __restrict__ negidx,
           float* __restrict__ out){
  __shared__ h16   Alds[CROWS][40];    // 128 rows x 32 k
  __shared__ h16   Blds[32][264];      // 32 k x 256 n of W1
  __shared__ float b1s[HID], w2s[HID];
  __shared__ float rowsum[CROWS];
  __shared__ float lossacc;

  const int tid  = threadIdx.x;
  const int wave = tid >> 5, lane = tid & 31;
  const int rowbase = blockIdx.x * CROWS;

  if (tid < HID){ b1s[tid] = b1[tid]; w2s[tid] = W2[tid]; }
  if (tid < CROWS) rowsum[tid] = 0.f;
  if (tid == 0) lossacc = 0.f;

  // row<8192 -> [anchor_i, pos_i]; row>=8192 -> [anchor_i, pos[negidx[i]]]
  const int lrow = tid >> 1;
  const int grow = rowbase + lrow;
  const int i0 = (grow < B_ROWS) ? grow : (grow - B_ROWS);
  const int nj = (grow < B_ROWS) ? i0   : negidx[grow - B_ROWS];
  const h16* srcA = xh + (size_t)i0 * FEAT;   // cols [0,512)
  const h16* srcB = xh + (size_t)nj * FEAT;   // cols [512,1024): x[nj][1][c-512]

  v8f acc[16];
  v8f vzero = {};
  #pragma unroll
  for (int t = 0; t < 16; ++t) acc[t] = vzero;

  for (int kb = 0; kb < FEAT; kb += 32){
    __syncthreads();
    { // feature tile: raw f16 copy, 32B per thread
      const int cg = (tid & 1) * 16;
      const int c = kb + cg;
      const h16* src = (c < DIM) ? (srcA + c) : (srcB + c);
      *(v8h*)&Alds[lrow][cg]     = *(const v8h*)src;
      *(v8h*)&Alds[lrow][cg + 8] = *(const v8h*)(src + 8);
    }
    { // W1 tile: raw f16 copy, 64B per thread
      const int k = tid >> 3, ng = (tid & 7) * 32;
      const h16* src = W1h + (size_t)(kb + k) * HID + ng;
      #pragma unroll
      for (int e = 0; e < 4; ++e)
        *(v8h*)&Blds[k][ng + e * 8] = *(const v8h*)(src + e * 8);
    }
    __syncthreads();
    v16h a = frag_a16(&Alds[wave * 16][0], 40);
    #pragma unroll
    for (int t = 0; t < 16; ++t){
      v16h bf = frag_b16(&Blds[0][0], 264, t * 16);
      acc[t] = __builtin_amdgcn_wmma_f32_16x16x32_f16(false, a, false, bf, (short)0, acc[t], false, false);
    }
  }
  __syncthreads();

  { // SiLU + dot(W2), reduce per row via LDS f32 atomics (ds_add_f32)
    const int n0 = lane & 15, mo = (lane < 16) ? 0 : 8;
    float part[8];
    #pragma unroll
    for (int r = 0; r < 8; ++r) part[r] = 0.f;
    #pragma unroll
    for (int t = 0; t < 16; ++t){
      const int n = t * 16 + n0;
      const float bn = b1s[n], wn = w2s[n];
      #pragma unroll
      for (int r = 0; r < 8; ++r){
        const float h = acc[t][r] + bn;
        const float s = h / (1.f + __expf(-h));   // SiLU
        part[r] += s * wn;
      }
    }
    #pragma unroll
    for (int r = 0; r < 8; ++r) atomicAdd(&rowsum[wave * 16 + r + mo], part[r]);
  }
  __syncthreads();

  if (tid < CROWS){
    const float z = rowsum[tid] + b2[0];
    float p = 1.f / (1.f + __expf(-z));
    p = fminf(fmaxf(p, 1e-7f), 1.f - 1e-7f);
    const int gr = rowbase + tid;
    const float term = (gr < B_ROWS) ? -__logf(p) : -__logf(1.f - p);
    atomicAdd(&lossacc, term);
  }
  __syncthreads();
  if (tid == 0) atomicAdd(out, lossacc * (1.f / (2.f * (float)B_ROWS)));
}

extern "C" void kernel_launch(void* const* d_in, const int* in_sizes, int n_in,
                              void* d_out, int out_size, void* d_ws, size_t ws_size,
                              hipStream_t stream){
  const float* x  = (const float*)d_in[0];
  const float* W1 = (const float*)d_in[1];
  const float* b1 = (const float*)d_in[2];
  const float* W2 = (const float*)d_in[3];
  const float* b2 = (const float*)d_in[4];
  float* out = (float*)d_out;

  float* cconst = (float*)d_ws;                                      // 8192 f32
  int*   negidx = (int*)((char*)d_ws + B_ROWS * sizeof(float));      // 8192 i32
  h16*   xh     = (h16*)((char*)d_ws + 65536);                       // 8192x1024 f16 (16 MB)
  h16*   W1h    = xh + (size_t)B_ROWS * FEAT;                        // 1024x256 f16 (512 KB)

  k_zero    <<<1, 64, 0, stream>>>(out);
  k_cvt_x   <<<B_ROWS, 256, 0, stream>>>(x, xh);
  k_cvt_w1  <<<(FEAT * HID) / 1024, 256, 0, stream>>>(W1, W1h);
  k_colconst<<<B_ROWS / 8, 256, 0, stream>>>(x, cconst);
  k_mine    <<<B_ROWS / MROWS, 128, 0, stream>>>(xh, cconst, negidx);
  k_classify<<<(2 * B_ROWS) / CROWS, 256, 0, stream>>>(xh, W1h, b1, W2, b2, negidx, out);
}
